// LinearAttention_46024869544196
// MI455X (gfx1250) — matrix operand order repaired and checked by reading, not compiled, not run on hardware
//
#include <hip/hip_runtime.h>

typedef __attribute__((ext_vector_type(16))) _Float16 v16h;
typedef __attribute__((ext_vector_type(8)))  _Float16 v8h;
typedef __attribute__((ext_vector_type(8)))  float    v8f;

#define B_DIM 2
#define C_DIM 128
#define N_DIM 8192   // 8*32*32
#define NBLK  (N_DIM / 32)

// Build a 16-half WMMA fragment from two contiguous 8-half (16B) chunks.
__device__ __forceinline__ v16h make_frag(const _Float16* p0, const _Float16* p1) {
  union { v16h v; v8h h[2]; } u;
  u.h[0] = *(const v8h*)p0;
  u.h[1] = *(const v8h*)p1;
  return u.v;
}

// CDNA5 async global -> LDS copy, 16B per lane, tracked by ASYNCcnt.
__device__ __forceinline__ void async_copy16(unsigned lds_off, unsigned long long gaddr) {
  asm volatile("global_load_async_to_lds_b128 %0, %1, off"
               :: "v"(lds_off), "v"(gaddr) : "memory");
}

// xor-butterfly across the 16 lanes of a half-wave using DPP (no LDS traffic).
constexpr int dpp8_sel_xor(int m) {
  int s = 0;
  for (int i = 0; i < 8; ++i) s |= ((i ^ m) & 7) << (3 * i);
  return s;
}
template <int M>
__device__ __forceinline__ float xor_bfly(float x) {
  int xi = __float_as_int(x);
  int r;
  if constexpr (M == 8) {
    r = __builtin_amdgcn_mov_dpp(xi, 0x128 /*row_ror:8*/, 0xf, 0xf, true);
  } else {
    r = __builtin_amdgcn_mov_dpp8(xi, dpp8_sel_xor(M));
  }
  return __int_as_float(r);
}
__device__ __forceinline__ float red_max16(float x) {
  x = fmaxf(x, xor_bfly<1>(x));
  x = fmaxf(x, xor_bfly<2>(x));
  x = fmaxf(x, xor_bfly<4>(x));
  x = fmaxf(x, xor_bfly<8>(x));
  return x;
}
__device__ __forceinline__ float red_sum16(float x) {
  x += xor_bfly<1>(x);
  x += xor_bfly<2>(x);
  x += xor_bfly<4>(x);
  x += xor_bfly<8>(x);
  return x;
}

// ---------------------------------------------------------------------------
// f32 [C][N] -> f16 [N][C] transpose+convert (LDS tiled)
// ---------------------------------------------------------------------------
__global__ void transpose_cvt_kernel(const float* __restrict__ src, _Float16* __restrict__ dst) {
  __shared__ _Float16 tile[32][33];
  const int n0 = blockIdx.x * 32, c0 = blockIdx.y * 32, b = blockIdx.z;
  src += (size_t)b * C_DIM * N_DIM;
  dst += (size_t)b * N_DIM * C_DIM;
  const int tx = threadIdx.x, ty = threadIdx.y;
#pragma unroll
  for (int i = 0; i < 4; ++i) {
    int c = c0 + ty + i * 8;
    tile[ty + i * 8][tx] = (_Float16)src[(size_t)c * N_DIM + n0 + tx];
  }
  __syncthreads();
#pragma unroll
  for (int i = 0; i < 4; ++i) {
    int n = n0 + ty + i * 8;
    dst[(size_t)n * C_DIM + c0 + tx] = tile[tx][ty + i * 8];
  }
}

// f32 -> f16 elementwise (weights)
__global__ void cvt_kernel(const float* __restrict__ s, _Float16* __restrict__ d, int n) {
  int i = blockIdx.x * blockDim.x + threadIdx.x;
  if (i < n) d[i] = (_Float16)s[i];
}

// ---------------------------------------------------------------------------
// 1x1x1 conv as WMMA GEMM:  D[o, n] = sum_c W[o,c] * srcT[n,c] + bias[o]
// MODE 0: dst f16 [N][C]; MODE 1: dst f16 [C][N]; MODE 2: dst f32 [C][N]
// ---------------------------------------------------------------------------
template <int MODE>
__global__ void __launch_bounds__(128) proj_kernel(const _Float16* __restrict__ W,
                                                   const float* __restrict__ bias,
                                                   const _Float16* __restrict__ srcT,
                                                   void* __restrict__ dstv) {
  const int tid  = threadIdx.x;
  const int wave = tid >> 5, lane = tid & 31;
  const int sub  = lane >> 4, l16 = lane & 15;
  const int n0 = blockIdx.x * 64 + wave * 16;
  const int o0 = blockIdx.y * 16;
  const int b  = blockIdx.z;
  srcT += (size_t)b * N_DIM * C_DIM;

  v8f acc = {};
#pragma unroll
  for (int cf = 0; cf < 4; ++cf) {
    const _Float16* wp = W + (size_t)(o0 + l16) * C_DIM + cf * 32 + sub * 8;
    v16h a = make_frag(wp, wp + 16);
    const _Float16* bp = srcT + (size_t)(n0 + l16) * C_DIM + cf * 32 + sub * 8;
    v16h bm = make_frag(bp, bp + 16);
    acc = __builtin_amdgcn_wmma_f32_16x16x32_f16(false, a, false, bm, (short)0, acc,
                                                 false, false);
  }

  if (MODE == 0) {
    _Float16* dst = (_Float16*)dstv + (size_t)b * N_DIM * C_DIM;
    v8h pk;
#pragma unroll
    for (int v = 0; v < 8; ++v) pk[v] = (_Float16)(acc[v] + bias[o0 + sub * 8 + v]);
    *(v8h*)(dst + (size_t)(n0 + l16) * C_DIM + o0 + sub * 8) = pk;
  } else if (MODE == 1) {
    _Float16* dst = (_Float16*)dstv + (size_t)b * C_DIM * N_DIM;
#pragma unroll
    for (int v = 0; v < 8; ++v)
      dst[(size_t)(o0 + sub * 8 + v) * N_DIM + n0 + l16] =
          (_Float16)(acc[v] + bias[o0 + sub * 8 + v]);
  } else {
    float* dst = (float*)dstv + (size_t)b * C_DIM * N_DIM;
#pragma unroll
    for (int v = 0; v < 8; ++v)
      dst[(size_t)(o0 + sub * 8 + v) * N_DIM + n0 + l16] = acc[v] + bias[o0 + sub * 8 + v];
  }
}

// ---------------------------------------------------------------------------
// Flash attention, 4 waves/block; K/V double-buffered in LDS via async
// global->LDS loads; DPP softmax reductions; exp2-domain online softmax;
// K fragments batch-loaded behind a sched_barrier for staged DS waits.
// ---------------------------------------------------------------------------
__global__ void __launch_bounds__(128) attn_kernel(const _Float16* __restrict__ qT,
                                                   const _Float16* __restrict__ kT,
                                                   const _Float16* __restrict__ vN,
                                                   _Float16* __restrict__ aoT) {
  __shared__ __align__(16) _Float16 ksh[2][32][136];   // K block: [key][channel]
  __shared__ __align__(16) _Float16 vsh[2][128][40];   // V block: [channel][key]
  __shared__ __align__(16) _Float16 pbuf[4][16][40];   // per-wave P relayout

  const int tid  = threadIdx.x;
  const int wave = tid >> 5, lane = tid & 31;
  const int sub  = lane >> 4, l16 = lane & 15;
  const int q0 = (blockIdx.x * 4 + wave) * 16;
  const int b  = blockIdx.y;
  qT  += (size_t)b * N_DIM * C_DIM;
  kT  += (size_t)b * N_DIM * C_DIM;
  vN  += (size_t)b * C_DIM * N_DIM;
  aoT += (size_t)b * N_DIM * C_DIM;

  auto stage = [&](int blk, int buf) {
    const int m0 = blk * 32;
#pragma unroll
    for (int j = 0; j < 4; ++j) {
      const int r = (tid >> 4) + j * 8;        // key row 0..31
      const int c8 = (tid & 15) * 8;           // channel chunk
      async_copy16((unsigned)(uintptr_t)&ksh[buf][r][c8],
                   (unsigned long long)(uintptr_t)(kT + (size_t)(m0 + r) * C_DIM + c8));
    }
#pragma unroll
    for (int j = 0; j < 4; ++j) {              // channel row = tid (0..127)
      async_copy16((unsigned)(uintptr_t)&vsh[buf][tid][j * 8],
                   (unsigned long long)(uintptr_t)(vN + (size_t)tid * N_DIM + m0 + j * 8));
    }
  };

  // Q fragments, pre-scaled by (1/sqrt(C)) * log2(e): softmax in exp2 domain.
  const _Float16 qs = (_Float16)0.1275174f;
  v16h aq[4];
#pragma unroll
  for (int cf = 0; cf < 4; ++cf) {
    const _Float16* qp = qT + (size_t)(q0 + l16) * C_DIM + cf * 32 + sub * 8;
    v16h a = make_frag(qp, qp + 16);
#pragma unroll
    for (int e = 0; e < 16; ++e) a[e] *= qs;
    aq[cf] = a;
  }

  v8f acc[8];
#pragma unroll
  for (int t = 0; t < 8; ++t) acc[t] = {};
  float m_run[8], l_run[8];
#pragma unroll
  for (int v = 0; v < 8; ++v) { m_run[v] = -1e30f; l_run[v] = 0.f; }

  stage(0, 0);

#pragma unroll 1
  for (int it = 0; it < NBLK; ++it) {
    const int buf = it & 1;
    if (it + 1 < NBLK) {
      stage(it + 1, buf ^ 1);
      asm volatile("s_wait_asynccnt 0x8" ::: "memory");  // block `it` landed
    } else {
      asm volatile("s_wait_asynccnt 0x0" ::: "memory");
    }
    __syncthreads();

    // Batch-issue ALL 16 ds_load_b128 for the 8 K fragments, then fence the
    // scheduler so the WMMAs stay below -> staged s_wait_dscnt, not 8 drains.
    v16h kf[8];
#pragma unroll
    for (int i = 0; i < 8; ++i) {
      const _Float16* p = &ksh[buf][(i & 1) * 16 + l16][(i >> 1) * 32 + sub * 8];
      kf[i] = make_frag(p, p + 16);
    }
    __builtin_amdgcn_sched_barrier(0);

    v8f s0 = {}, s1 = {};
#pragma unroll
    for (int cf = 0; cf < 4; ++cf) {
      s0 = __builtin_amdgcn_wmma_f32_16x16x32_f16(false, aq[cf], false, kf[2 * cf], (short)0,
                                                  s0, false, false);
      s1 = __builtin_amdgcn_wmma_f32_16x16x32_f16(false, aq[cf], false, kf[2 * cf + 1], (short)0,
                                                  s1, false, false);
    }

    // Prefetch V fragments; fence so the softmax can't be hoisted above the
    // load issue -> their DS latency hides under the softmax VALU work.
    v16h vf[8];
#pragma unroll
    for (int t = 0; t < 8; ++t) {
      const _Float16* vp = &vsh[buf][t * 16 + l16][sub * 8];
      vf[t] = make_frag(vp, vp + 16);
    }
    __builtin_amdgcn_sched_barrier(0);

    // Online softmax in exp2 domain; rows span 16 lanes -> DPP reductions.
    float scl[8];
#pragma unroll
    for (int v = 0; v < 8; ++v) {
      float a0 = s0[v], a1 = s1[v];
      float m = red_max16(fmaxf(a0, a1));
      float mn = fmaxf(m_run[v], m);
      float p0 = exp2f(a0 - mn), p1 = exp2f(a1 - mn);
      float l = red_sum16(p0 + p1);
      scl[v]   = exp2f(m_run[v] - mn);
      l_run[v] = l_run[v] * scl[v] + l;
      m_run[v] = mn;
      s0[v] = p0; s1[v] = p1;
    }
#pragma unroll
    for (int t = 0; t < 8; ++t)
#pragma unroll
      for (int v = 0; v < 8; ++v) acc[t][v] *= scl[v];

    // Relay P (C/D layout) -> A-fragment layout via per-wave LDS.
#pragma unroll
    for (int v = 0; v < 8; ++v) {
      pbuf[wave][sub * 8 + v][l16]      = (_Float16)s0[v];
      pbuf[wave][sub * 8 + v][16 + l16] = (_Float16)s1[v];
    }
    v16h pf = make_frag(&pbuf[wave][l16][sub * 8], &pbuf[wave][l16][16 + sub * 8]);

    // O += P * V^T
#pragma unroll
    for (int t = 0; t < 8; ++t)
      acc[t] = __builtin_amdgcn_wmma_f32_16x16x32_f16(false, pf, false, vf[t], (short)0,
                                                      acc[t], false, false);
    __syncthreads();  // buf may be overwritten by next iteration's prefetch
  }

  // Normalize and store aoT f16 [N][C]
#pragma unroll
  for (int v = 0; v < 8; ++v) {
    float inv = 1.0f / l_run[v];
#pragma unroll
    for (int t = 0; t < 8; ++t)
      aoT[(size_t)(q0 + sub * 8 + v) * C_DIM + t * 16 + l16] = (_Float16)(acc[t][v] * inv);
  }
}

// ---------------------------------------------------------------------------
extern "C" void kernel_launch(void* const* d_in, const int* in_sizes, int n_in,
                              void* d_out, int out_size, void* d_ws, size_t ws_size,
                              hipStream_t stream) {
  (void)in_sizes; (void)n_in; (void)out_size; (void)ws_size;
  const float* x    = (const float*)d_in[0];
  const float* h    = (const float*)d_in[1];
  const float* Wq_w = (const float*)d_in[2];
  const float* Wq_b = (const float*)d_in[3];
  const float* Wk_w = (const float*)d_in[4];
  const float* Wk_b = (const float*)d_in[5];
  const float* Wv_w = (const float*)d_in[6];
  const float* Wv_b = (const float*)d_in[7];
  const float* Wo_w = (const float*)d_in[8];
  const float* Wo_b = (const float*)d_in[9];

  char* ws = (char*)d_ws;
  size_t off = 0;
  auto alloc = [&](size_t bytes) -> char* {
    char* p = ws + off;
    off += (bytes + 255) & ~(size_t)255;
    return p;
  };
  const size_t big = (size_t)B_DIM * N_DIM * C_DIM * sizeof(_Float16);
  _Float16* xT   = (_Float16*)alloc(big);
  _Float16* hT   = (_Float16*)alloc(big);
  _Float16* qT   = (_Float16*)alloc(big);
  _Float16* kT   = (_Float16*)alloc(big);
  _Float16* vNw  = (_Float16*)alloc(big);
  _Float16* aoT  = (_Float16*)alloc(big);
  _Float16* Wq16 = (_Float16*)alloc((size_t)C_DIM * C_DIM * 2);
  _Float16* Wk16 = (_Float16*)alloc((size_t)C_DIM * C_DIM * 2);
  _Float16* Wv16 = (_Float16*)alloc((size_t)C_DIM * C_DIM * 2);
  _Float16* Wo16 = (_Float16*)alloc((size_t)C_DIM * C_DIM * 2);

  dim3 tb(32, 8);
  dim3 tg(N_DIM / 32, C_DIM / 32, B_DIM);
  transpose_cvt_kernel<<<tg, tb, 0, stream>>>(x, xT);
  transpose_cvt_kernel<<<tg, tb, 0, stream>>>(h, hT);

  const int wn = C_DIM * C_DIM;
  cvt_kernel<<<(wn + 255) / 256, 256, 0, stream>>>(Wq_w, Wq16, wn);
  cvt_kernel<<<(wn + 255) / 256, 256, 0, stream>>>(Wk_w, Wk16, wn);
  cvt_kernel<<<(wn + 255) / 256, 256, 0, stream>>>(Wv_w, Wv16, wn);
  cvt_kernel<<<(wn + 255) / 256, 256, 0, stream>>>(Wo_w, Wo16, wn);

  dim3 pg(N_DIM / 64, C_DIM / 16, B_DIM);
  proj_kernel<0><<<pg, 128, 0, stream>>>(Wq16, Wq_b, xT, qT);
  proj_kernel<0><<<pg, 128, 0, stream>>>(Wk16, Wk_b, hT, kT);
  proj_kernel<1><<<pg, 128, 0, stream>>>(Wv16, Wv_b, hT, vNw);

  attn_kernel<<<dim3(N_DIM / 64, B_DIM), 128, 0, stream>>>(qT, kT, vNw, aoT);

  proj_kernel<2><<<pg, 128, 0, stream>>>(Wo16, Wo_b, aoT, d_out);
}
